// DeformableEncoderLayer_86199993630776
// MI455X (gfx1250) — matrix operand order repaired and checked by reading, not compiled, not run on hardware
//
#include <hip/hip_runtime.h>
#include <hip/hip_bf16.h>
#include <math.h>

// ---------------- problem constants (from reference) ----------------
#define D_   256
#define NH_  8
#define DH_  32
#define NL_  4
#define NP_  4
#define FFN_ 2048
#define EPS_ 1e-5f
#define BATCH_ 4

typedef __bf16 bf16_t;
typedef __attribute__((ext_vector_type(8)))  __bf16 v8bf;
typedef __attribute__((ext_vector_type(16))) __bf16 v16bf;
typedef __attribute__((ext_vector_type(8)))  float  v8f;

// ---------------- wave32 reductions ----------------
__device__ __forceinline__ float wave_sum32(float v) {
#pragma unroll
    for (int m = 16; m >= 1; m >>= 1) v += __shfl_xor(v, m, 32);
    return v;
}
__device__ __forceinline__ float wave_max32(float v) {
#pragma unroll
    for (int m = 16; m >= 1; m >>= 1) v = fmaxf(v, __shfl_xor(v, m, 32));
    return v;
}

// ---------------- weight convert + transpose: f32 [K][N] -> bf16 [N][K] ----------------
__global__ __launch_bounds__(256) void wt_to_bf16_t(const float* __restrict__ w,
                                                    bf16_t* __restrict__ wt,
                                                    int K, int N) {
    int i = blockIdx.x * 256 + threadIdx.x;
    if (i >= K * N) return;
    int k = i / N, n = i - k * N;
    wt[(size_t)n * K + k] = (bf16_t)w[i];
}

// ---------------- LN1: q = LN(x)*g+b + pos (bf16), v = x + pos (bf16) ----------------
__global__ __launch_bounds__(256) void ln1_kernel(const float* __restrict__ x,
                                                  const float* __restrict__ pos,
                                                  const float* __restrict__ g,
                                                  const float* __restrict__ b,
                                                  bf16_t* __restrict__ qb,
                                                  bf16_t* __restrict__ vb, int M) {
    const int lane = threadIdx.x & 31;
    const int t = blockIdx.x * 8 + (threadIdx.x >> 5);
    if (t >= M) return;
    const float* xr = x + (size_t)t * D_;
    const float* pr = pos + (size_t)t * D_;
    float vals[8];
    float s = 0.f;
#pragma unroll
    for (int j = 0; j < 8; ++j) { vals[j] = xr[j * 32 + lane]; s += vals[j]; }
    const float mu = wave_sum32(s) * (1.f / (float)D_);
    float s2 = 0.f;
#pragma unroll
    for (int j = 0; j < 8; ++j) { float d = vals[j] - mu; s2 += d * d; }
    const float rstd = rsqrtf(wave_sum32(s2) * (1.f / (float)D_) + EPS_);
#pragma unroll
    for (int j = 0; j < 8; ++j) {
        int c = j * 32 + lane;
        float p = pr[c];
        qb[(size_t)t * D_ + c] = (bf16_t)((vals[j] - mu) * rstd * g[c] + b[c] + p);
        vb[(size_t)t * D_ + c] = (bf16_t)(vals[j] + p);
    }
}

// ---------------- LN2: y = LN(x2)*g+b (bf16) ----------------
__global__ __launch_bounds__(256) void ln2_kernel(const float* __restrict__ x,
                                                  const float* __restrict__ g,
                                                  const float* __restrict__ b,
                                                  bf16_t* __restrict__ yb, int M) {
    const int lane = threadIdx.x & 31;
    const int t = blockIdx.x * 8 + (threadIdx.x >> 5);
    if (t >= M) return;
    const float* xr = x + (size_t)t * D_;
    float vals[8];
    float s = 0.f;
#pragma unroll
    for (int j = 0; j < 8; ++j) { vals[j] = xr[j * 32 + lane]; s += vals[j]; }
    const float mu = wave_sum32(s) * (1.f / (float)D_);
    float s2 = 0.f;
#pragma unroll
    for (int j = 0; j < 8; ++j) { float d = vals[j] - mu; s2 += d * d; }
    const float rstd = rsqrtf(wave_sum32(s2) * (1.f / (float)D_) + EPS_);
#pragma unroll
    for (int j = 0; j < 8; ++j) {
        int c = j * 32 + lane;
        yb[(size_t)t * D_ + c] = (bf16_t)((vals[j] - mu) * rstd * g[c] + b[c]);
    }
}

// ---------------- WMMA GEMM: C[M,N] = A[M,K](bf16) @ Bt[N,K](bf16)^T + bias ----------------
// One wave computes a 32x64 tile: 2 A fragments x 4 B fragments = 8 WMMAs per K-step.
// B fragments are reused across both M-subtiles (1.5 b128 loads per WMMA).
// EPI: 0 = store f32
//      1 = store f32, zero rows where mask[m]!=0          (value projection)
//      2 = store f32, += res[m,n]                          (residual epilogues)
//      3 = exact GELU, store bf16                          (FFN up-proj)
template <int EPI>
__global__ __launch_bounds__(256) void gemm_wmma(const bf16_t* __restrict__ A,
                                                 const bf16_t* __restrict__ Bt,
                                                 const float* __restrict__ bias,
                                                 const float* __restrict__ res,
                                                 const unsigned char* __restrict__ mask,
                                                 void* __restrict__ Cout,
                                                 int M, int N, int K) {
    const int lane = threadIdx.x & 31;
    const int gw = blockIdx.x * 8 + (threadIdx.x >> 5);
    const int tiles_n = N >> 6;                 // 64-wide wave tiles
    const int tiles_m = M >> 5;                 // 32-tall wave tiles
    if (gw >= tiles_m * tiles_n) return;
    const int m0 = (gw / tiles_n) << 5;
    const int tn = (gw % tiles_n) << 6;

    const int half = lane >> 4;                 // 0: K 0..7 / 16..23,  1: K 8..15 / 24..31
    const int r = lane & 15;                    // A row within tile / B,C column within subtile

    v8f acc[2][4] = {};
    const bf16_t* pa0 = A + (size_t)(m0 + r) * K;
    const bf16_t* pa1 = A + (size_t)(m0 + 16 + r) * K;

    for (int k0 = 0; k0 < K; k0 += 32) {
        // A fragments (16-bit A 16x32 layout): two contiguous 8-element chunks per lane
        v8bf a0_lo = *(const v8bf*)(pa0 + k0 + 8 * half);
        v8bf a0_hi = *(const v8bf*)(pa0 + k0 + 16 + 8 * half);
        v16bf af0 = __builtin_shufflevector(a0_lo, a0_hi, 0, 1, 2, 3, 4, 5, 6, 7,
                                            8, 9, 10, 11, 12, 13, 14, 15);
        v8bf a1_lo = *(const v8bf*)(pa1 + k0 + 8 * half);
        v8bf a1_hi = *(const v8bf*)(pa1 + k0 + 16 + 8 * half);
        v16bf af1 = __builtin_shufflevector(a1_lo, a1_hi, 0, 1, 2, 3, 4, 5, 6, 7,
                                            8, 9, 10, 11, 12, 13, 14, 15);
        __builtin_prefetch(pa0 + k0 + 64, 0, 1);   // -> global_prefetch_b8 (stream next A)
        __builtin_prefetch(pa1 + k0 + 64, 0, 1);
#pragma unroll
        for (int s = 0; s < 4; ++s) {
            // B fragment (32x16 layout): lanes 0-15 K=0..7/16..23, lanes 16-31 K=8..15/24..31, N = r
            const bf16_t* pb = Bt + (size_t)(tn + s * 16 + r) * K + k0 + 8 * half;
            v8bf b_lo = *(const v8bf*)(pb);
            v8bf b_hi = *(const v8bf*)(pb + 16);
            v16bf bfr = __builtin_shufflevector(b_lo, b_hi, 0, 1, 2, 3, 4, 5, 6, 7,
                                                8, 9, 10, 11, 12, 13, 14, 15);
            acc[0][s] = __builtin_amdgcn_wmma_f32_16x16x32_bf16(
                false, af0, false, bfr, (short)0, acc[0][s], false, false);
            acc[1][s] = __builtin_amdgcn_wmma_f32_16x16x32_bf16(
                false, af1, false, bfr, (short)0, acc[1][s], false, false);
        }
    }

    // C/D layout: acc[mi][s][rr] holds (m = m0 + 16*mi + 8*half + rr, n = tn + s*16 + r)
#pragma unroll
    for (int mi = 0; mi < 2; ++mi) {
#pragma unroll
        for (int s = 0; s < 4; ++s) {
            const int n = tn + s * 16 + r;
            const float bv = bias[n];
#pragma unroll
            for (int rr = 0; rr < 8; ++rr) {
                const int m = m0 + 16 * mi + 8 * half + rr;
                float v = acc[mi][s][rr] + bv;
                if (EPI == 1) v = mask[m] ? 0.f : v;
                if (EPI == 2) v += res[(size_t)m * N + n];
                if (EPI == 3) {
                    v = 0.5f * v * (1.f + erff(v * 0.70710678118654752f));
                    ((bf16_t*)Cout)[(size_t)m * N + n] = (bf16_t)v;
                } else {
                    ((float*)Cout)[(size_t)m * N + n] = v;
                }
            }
        }
    }
}

// ---------------- deformable sampling: one wave per (token, head), lane = channel ----------------
__global__ __launch_bounds__(256) void deform_sample(const float* __restrict__ value,
                                                     const float* __restrict__ off,
                                                     const float* __restrict__ logits,
                                                     const float* __restrict__ ref,
                                                     const int* __restrict__ shapes,
                                                     bf16_t* __restrict__ samp, int Lq) {
    const int lane = threadIdx.x & 31;
    const int wid = blockIdx.x * 8 + (threadIdx.x >> 5);
    const int Mtot = BATCH_ * Lq;
    const int t = wid >> 3;        // token
    const int hd = wid & 7;        // head
    if (t >= Mtot) return;
    const int b = t / Lq;

    int Hs[NL_], Ws[NL_], starts[NL_];
    int a0 = 0;
#pragma unroll
    for (int l = 0; l < NL_; ++l) {
        Hs[l] = shapes[2 * l];
        Ws[l] = shapes[2 * l + 1];
        starts[l] = a0;
        a0 += Hs[l] * Ws[l];
    }

    // softmax over 32 attention logits (one per lane; NL*NP is 16)
    const float* lg = logits + ((size_t)t * NH_ + hd) * (NL_ * NP_);
    float logit = lg[lane];
    float mx = wave_max32(logit);
    float e = __expf(logit - mx);
    float aw = e / wave_sum32(e);

    // lane p holds (dx, dy) of point p
    const float* ofb = off + ((size_t)t * NH_ + hd) * (NL_ * NP_ * 2);
    float offx = ofb[2 * lane];
    float offy = ofb[2 * lane + 1];

    const float* refb = ref + (size_t)t * NL_ * 2;
    const float* vbase = value + (size_t)b * Lq * D_ + hd * DH_ + lane;

    float accv = 0.f;
#pragma unroll
    for (int l = 0; l < NL_; ++l) {
        const int Wl = Ws[l], Hl = Hs[l], st = starts[l];
        const float Wf = (float)Wl, Hf = (float)Hl;
        const float rx = refb[2 * l], ry = refb[2 * l + 1];
#pragma unroll
        for (int p = 0; p < NP_; ++p) {
            const int pt = l * NP_ + p;
            const float dx = __shfl(offx, pt, 32);
            const float dy = __shfl(offy, pt, 32);
            const float a_w = __shfl(aw, pt, 32);
            const float xs = (rx + dx / Wf) * Wf - 0.5f;
            const float ys = (ry + dy / Hf) * Hf - 0.5f;
            const float x0f = floorf(xs), y0f = floorf(ys);
            const int x0 = (int)x0f, y0 = (int)y0f;
            const float wx = xs - x0f, wy = ys - y0f;
#pragma unroll
            for (int c = 0; c < 4; ++c) {
                const int ix = x0 + (c & 1);
                const int iy = y0 + (c >> 1);
                float w = ((c & 1) ? wx : 1.f - wx) * ((c >> 1) ? wy : 1.f - wy);
                const bool valid = (ix >= 0) && (ix < Wl) && (iy >= 0) && (iy < Hl);
                const int cx = min(max(ix, 0), Wl - 1);
                const int cy = min(max(iy, 0), Hl - 1);
                const float gv = vbase[(size_t)(st + cy * Wl + cx) * D_];
                accv += a_w * (valid ? w : 0.f) * gv;
            }
        }
    }
    samp[(size_t)t * D_ + hd * DH_ + lane] = (bf16_t)accv;
}

// ---------------- host-side orchestration ----------------
static inline int gemm_blocks(int M, int N) { return ((M >> 5) * (N >> 6) + 7) / 8; }

extern "C" void kernel_launch(void* const* d_in, const int* in_sizes, int n_in,
                              void* d_out, int out_size, void* d_ws, size_t ws_size,
                              hipStream_t stream) {
    const float* x      = (const float*)d_in[0];
    const float* ref    = (const float*)d_in[1];
    const int*   ss     = (const int*)d_in[2];
    const unsigned char* mask = (const unsigned char*)d_in[3];
    const float* pos    = (const float*)d_in[4];
    const float* w_off  = (const float*)d_in[5];
    const float* b_off  = (const float*)d_in[6];
    const float* w_attn = (const float*)d_in[7];
    const float* b_attn = (const float*)d_in[8];
    const float* w_val  = (const float*)d_in[9];
    const float* b_val  = (const float*)d_in[10];
    const float* w_out  = (const float*)d_in[11];
    const float* b_out  = (const float*)d_in[12];
    const float* ln1g   = (const float*)d_in[13];
    const float* ln1b   = (const float*)d_in[14];
    const float* ln2g   = (const float*)d_in[15];
    const float* ln2b   = (const float*)d_in[16];
    const float* w1     = (const float*)d_in[17];
    const float* b1     = (const float*)d_in[18];
    const float* w2     = (const float*)d_in[19];
    const float* b2     = (const float*)d_in[20];

    const int Lq = in_sizes[0] / (BATCH_ * D_);
    const int M = BATCH_ * Lq;
    const int NOFF = NH_ * NL_ * NP_ * 2;   // 256
    const int NATT = NH_ * NL_ * NP_;       // 128

    // workspace carve-out (256B aligned slices)
    char* ws = (char*)d_ws;
    size_t cur = 0;
    auto take = [&](size_t bytes) -> char* {
        char* p = ws + cur;
        cur += (bytes + 255) & ~(size_t)255;
        return p;
    };
    bf16_t* q_b    = (bf16_t*)take((size_t)M * D_ * 2);
    bf16_t* v_b    = (bf16_t*)take((size_t)M * D_ * 2);
    float*  valuef = (float*)take((size_t)M * D_ * 4);
    float*  offv   = (float*)take((size_t)M * NOFF * 4);
    float*  attnlg = (float*)take((size_t)M * NATT * 4);
    bf16_t* sampb  = (bf16_t*)take((size_t)M * D_ * 2);
    float*  x2     = (float*)take((size_t)M * D_ * 4);
    bf16_t* y_b    = (bf16_t*)take((size_t)M * D_ * 2);
    bf16_t* h_b    = (bf16_t*)take((size_t)M * FFN_ * 2);
    bf16_t* wvalT  = (bf16_t*)take((size_t)D_ * D_ * 2);
    bf16_t* woffT  = (bf16_t*)take((size_t)D_ * NOFF * 2);
    bf16_t* wattnT = (bf16_t*)take((size_t)D_ * NATT * 2);
    bf16_t* woutT  = (bf16_t*)take((size_t)D_ * D_ * 2);
    bf16_t* w1T    = (bf16_t*)take((size_t)D_ * FFN_ * 2);
    bf16_t* w2T    = (bf16_t*)take((size_t)FFN_ * D_ * 2);
    (void)ws_size;

    // 1) weights -> bf16, transposed to [N][K]
    wt_to_bf16_t<<<(D_ * D_ + 255) / 256, 256, 0, stream>>>(w_val, wvalT, D_, D_);
    wt_to_bf16_t<<<(D_ * NOFF + 255) / 256, 256, 0, stream>>>(w_off, woffT, D_, NOFF);
    wt_to_bf16_t<<<(D_ * NATT + 255) / 256, 256, 0, stream>>>(w_attn, wattnT, D_, NATT);
    wt_to_bf16_t<<<(D_ * D_ + 255) / 256, 256, 0, stream>>>(w_out, woutT, D_, D_);
    wt_to_bf16_t<<<(D_ * FFN_ + 255) / 256, 256, 0, stream>>>(w1, w1T, D_, FFN_);
    wt_to_bf16_t<<<(FFN_ * D_ + 255) / 256, 256, 0, stream>>>(w2, w2T, FFN_, D_);

    // 2) LN1 -> q (bf16), v = x + pos (bf16)
    ln1_kernel<<<(M + 7) / 8, 256, 0, stream>>>(x, pos, ln1g, ln1b, q_b, v_b, M);

    // 3) projections on matrix cores
    gemm_wmma<1><<<gemm_blocks(M, D_), 256, 0, stream>>>(v_b, wvalT, b_val, nullptr, mask,
                                                         valuef, M, D_, D_);
    gemm_wmma<0><<<gemm_blocks(M, NOFF), 256, 0, stream>>>(q_b, woffT, b_off, nullptr, nullptr,
                                                           offv, M, NOFF, D_);
    gemm_wmma<0><<<gemm_blocks(M, NATT), 256, 0, stream>>>(q_b, wattnT, b_attn, nullptr, nullptr,
                                                           attnlg, M, NATT, D_);

    // 4) softmax + bilinear deformable sampling (one wave per token-head)
    deform_sample<<<(M * NH_ + 7) / 8, 256, 0, stream>>>(valuef, offv, attnlg, ref, ss, sampb, Lq);

    // 5) output projection + residual -> x2
    gemm_wmma<2><<<gemm_blocks(M, D_), 256, 0, stream>>>(sampb, woutT, b_out, x, nullptr,
                                                         x2, M, D_, D_);

    // 6) LN2 -> y (bf16)
    ln2_kernel<<<(M + 7) / 8, 256, 0, stream>>>(x2, ln2g, ln2b, y_b, M);

    // 7) FFN: GELU(y@w1+b1) (bf16) then @w2+b2 + x2 -> d_out
    gemm_wmma<3><<<gemm_blocks(M, FFN_), 256, 0, stream>>>(y_b, w1T, b1, nullptr, nullptr,
                                                           h_b, M, FFN_, D_);
    gemm_wmma<2><<<gemm_blocks(M, D_), 256, 0, stream>>>(h_b, w2T, b2, x2, nullptr,
                                                         (float*)d_out, M, D_, FFN_);
}
